// BlockRecurrentTransformerWrapper_76579266888052
// MI455X (gfx1250) — matrix operand order repaired and checked by reading, not compiled, not run on hardware
//
#include <hip/hip_runtime.h>

// ---------------------------------------------------------------------------
// Block-recurrent transformer forward for MI455X (gfx1250, wave32, WMMA).
// All matmuls -> v_wmma_f32_16x16x32_bf16 (f32 accum). Weights transposed to
// bf16 [N,K] once per launch; activations kept bf16 between GEMMs; fused
// epilogues (residual, gelu, sigmoid-EMA state update).
// B tiles staged in LDS with a double-buffered ASYNC global->LDS pipeline
// (GLOBAL_LOAD_ASYNC_TO_LDS_B128 + s_wait_asynccnt) when the toolchain
// exposes the builtins; falls back to load+ds_store otherwise.
// ---------------------------------------------------------------------------

typedef __attribute__((ext_vector_type(16))) __bf16      v16bf;
typedef __attribute__((ext_vector_type(8)))  float       v8f;
typedef __attribute__((ext_vector_type(4)))  unsigned int v4u;
typedef int v4i __attribute__((__vector_size__(16)));   // matches async builtin param

typedef unsigned short us;

#define DEV __device__ __forceinline__

#if defined(__has_builtin)
#if __has_builtin(__builtin_amdgcn_global_load_async_to_lds_b128) && \
    __has_builtin(__builtin_amdgcn_s_wait_asynccnt)
#define USE_ASYNC_LDS 1
#endif
#endif
#ifndef USE_ASYNC_LDS
#define USE_ASYNC_LDS 0
#endif

static constexpr int Bc = 4, Lc = 4096, Dc = 512, Hc = 8, DHc = 64;
static constexpr int NLc = 4, FFc = 2048, Sc = 512, Wc = 512, Ic = 64, Oc = 64;
static constexpr int NBc = Lc / Wc;   // 8 chunks
static constexpr int RECc = 1;

DEV us f2bf(float f) {
  unsigned u = __float_as_uint(f);
  unsigned r = u + 0x7FFFu + ((u >> 16) & 1u);   // round-to-nearest-even
  return (us)(r >> 16);
}
DEV float bf2f(us h) { return __uint_as_float(((unsigned)h) << 16); }

#define GSL(i, n) for (long long i = (long long)blockIdx.x * blockDim.x + threadIdx.x; \
                       i < (n); i += (long long)gridDim.x * blockDim.x)

// ---------------------------------------------------------------------------
// WMMA GEMM: C[M,N] = A[M,K](bf16,row-major) * Bt[N,K](bf16, i.e. B^T).
// Block 256 thr = 8 wave32, each wave a 16x64 strip -> 128x64 C tile/block.
// B tile (64x32 bf16 = 4KB) double-buffered in LDS, shared across waves.
// Batched over grid.z with two-level strides: off = (z/bdiv)*so + (z%bdiv)*si.
// EPI: 0=f32 store, 1=bf16 store, 2=f32 res+acc, 3=bf16 gelu(acc),
//      4=f32 sigmoid(beta)*old + (1-sig)*acc  (EMA state update)
// ---------------------------------------------------------------------------
struct GemmArgs {
  const us* A; const us* Bt;
  float* C; us* Cb;
  const float* res; const float* stold; const float* beta;
  long long soA, siA, soB, siB, soC, siC;
  int M, N, K, lda, ldb, ldc, bdiv;
};

#define KSTEP 32

template <int EPI>
__global__ __launch_bounds__(256) void k_gemm(GemmArgs p) {
  __shared__ __align__(16) us Bs[2][64 * KSTEP];   // 2 x (64 cols x 32 k) = 8KB

  const int tid  = threadIdx.x;
  const int wave = tid >> 5;
  const int lane = tid & 31;
  const int l16  = lane & 15;
  const int hi   = lane >> 4;

  const int z = blockIdx.z;
  const long long offA = (long long)(z / p.bdiv) * p.soA + (long long)(z % p.bdiv) * p.siA;
  const long long offB = (long long)(z / p.bdiv) * p.soB + (long long)(z % p.bdiv) * p.siB;
  const long long offC = (long long)(z / p.bdiv) * p.soC + (long long)(z % p.bdiv) * p.siC;

  const us* A  = p.A  + offA;
  const us* Bt = p.Bt + offB;

  const int m0 = blockIdx.y * 128 + wave * 16;
  const int n0 = blockIdx.x * 64;

  const int arow = m0 + l16;
  const int kh = hi ? 8 : 0;     // A frag k-half within 32
  const int kb = hi ? 8 : 0;     // B frag k-half within 32

  v8f acc[4];
#pragma unroll
  for (int nt = 0; nt < 4; ++nt)
#pragma unroll
    for (int r = 0; r < 8; ++r) acc[nt][r] = 0.f;

  union Frag { v4u u[2]; v16bf v; };

  const int brow = tid >> 2;            // 0..63  (tile column index n)
  const int bcol = (tid & 3) * 8;       // 0,8,16,24 (k chunk)
  const int lofs = brow * KSTEP + bcol; // LDS element offset within a buffer

  // stage the B tile for k-step k0 into LDS buffer `buf`
  auto stage = [&](int buf, int k0) {
    const us* gp = Bt + (long long)(n0 + brow) * p.ldb + k0 + bcol;
#if USE_ASYNC_LDS
    __builtin_amdgcn_global_load_async_to_lds_b128(
        (__attribute__((address_space(1))) v4i*)(const_cast<us*>(gp)),
        (__attribute__((address_space(3))) v4i*)(&Bs[buf][lofs]),
        0, 0);
#else
    *(v4u*)&Bs[buf][lofs] = *(const v4u*)gp;
#endif
  };

  const int nk = p.K / KSTEP;
  stage(0, 0);   // prologue fill of buffer 0

  for (int kt = 0; kt < nk; ++kt) {
    const int k0 = kt * KSTEP;
    // protect the buffer we are about to refill (read during iter kt-1)
    __syncthreads();
    if (kt + 1 < nk) {
      stage((kt + 1) & 1, k0 + KSTEP);
#if USE_ASYNC_LDS
      __builtin_amdgcn_s_wait_asynccnt(1);   // current tile complete, next in flight
#endif
    } else {
#if USE_ASYNC_LDS
      __builtin_amdgcn_s_wait_asynccnt(0);
#endif
    }
    __syncthreads();   // tile visible to all waves

    Frag af;
    const us* ap = A + (long long)arow * p.lda + k0 + kh;
    af.u[0] = *(const v4u*)(ap);
    af.u[1] = *(const v4u*)(ap + 16);
    if (k0 + KSTEP < p.K) __builtin_prefetch(ap + KSTEP, 0, 0);   // global_prefetch_b8

    const us* bbuf = Bs[kt & 1];
#pragma unroll
    for (int nt = 0; nt < 4; ++nt) {
      Frag bf_;
      const us* bp = bbuf + (nt * 16 + l16) * KSTEP + kb;
      bf_.u[0] = *(const v4u*)(bp);
      bf_.u[1] = *(const v4u*)(bp + 16);
      acc[nt] = __builtin_amdgcn_wmma_f32_16x16x32_bf16(
          false, af.v, false, bf_.v, (short)0, acc[nt], false, false);
    }
  }

  const int rbase = m0 + (hi ? 8 : 0);
#pragma unroll
  for (int nt = 0; nt < 4; ++nt) {
    const int col = n0 + nt * 16 + l16;
#pragma unroll
    for (int r = 0; r < 8; ++r) {
      const long long ci = offC + (long long)(rbase + r) * p.ldc + col;
      float v = acc[nt][r];
      if constexpr (EPI == 0) {
        p.C[ci] = v;
      } else if constexpr (EPI == 1) {
        p.Cb[ci] = f2bf(v);
      } else if constexpr (EPI == 2) {
        p.C[ci] = p.res[ci] + v;
      } else if constexpr (EPI == 3) {
        float g = 0.5f * v * (1.f + tanhf(0.7978845608f * (v + 0.044715f * v * v * v)));
        p.Cb[ci] = f2bf(g);
      } else if constexpr (EPI == 4) {
        float dcy = 1.f / (1.f + __expf(-p.beta[col]));
        p.C[ci] = dcy * p.stold[ci] + (1.f - dcy) * v;
      }
    }
  }
}

// ---------------------------------------------------------------------------
// LayerNorm: one block per row, f32 in -> bf16 out (feeds WMMA A operand).
// ---------------------------------------------------------------------------
__global__ __launch_bounds__(256) void k_ln(const float* __restrict__ x,
                                            const float* __restrict__ g,
                                            us* __restrict__ out, int D) {
  __shared__ float red[256];
  const long long row = blockIdx.x;
  const float* xr = x + row * D;
  float s = 0.f;
  for (int i = threadIdx.x; i < D; i += 256) s += xr[i];
  red[threadIdx.x] = s; __syncthreads();
  for (int o = 128; o > 0; o >>= 1) { if (threadIdx.x < o) red[threadIdx.x] += red[threadIdx.x + o]; __syncthreads(); }
  const float mean = red[0] / D; __syncthreads();
  float v = 0.f;
  for (int i = threadIdx.x; i < D; i += 256) { float d = xr[i] - mean; v += d * d; }
  red[threadIdx.x] = v; __syncthreads();
  for (int o = 128; o > 0; o >>= 1) { if (threadIdx.x < o) red[threadIdx.x] += red[threadIdx.x + o]; __syncthreads(); }
  const float rstd = rsqrtf(red[0] / D + 1e-5f);
  for (int i = threadIdx.x; i < D; i += 256)
    out[row * D + i] = f2bf((xr[i] - mean) * rstd * g[i]);
}

// ---------------------------------------------------------------------------
// Softmax with optional block-causal mask; f32 scores -> bf16 probs.
// grid = (rows, batches). mask_mode 1: allowed = (j <= row+W) && (j>=W || bidx>0)
// ---------------------------------------------------------------------------
__global__ __launch_bounds__(256) void k_softmax(const float* __restrict__ S,
                                                 us* __restrict__ P, int cols,
                                                 float scale, int mask_mode,
                                                 int bidx, int Wq) {
  __shared__ float red[256];
  const int row = blockIdx.x;
  const long long base = ((long long)blockIdx.y * gridDim.x + row) * cols;
  float mx = -3.0e38f;
  for (int j = threadIdx.x; j < cols; j += 256) {
    float v = S[base + j] * scale;
    if (mask_mode == 1 && !((j <= row + Wq) && (j >= Wq || bidx > 0))) v = -1e30f;
    mx = fmaxf(mx, v);
  }
  red[threadIdx.x] = mx; __syncthreads();
  for (int o = 128; o > 0; o >>= 1) { if (threadIdx.x < o) red[threadIdx.x] = fmaxf(red[threadIdx.x], red[threadIdx.x + o]); __syncthreads(); }
  mx = red[0]; __syncthreads();
  float sum = 0.f;
  for (int j = threadIdx.x; j < cols; j += 256) {
    float v = S[base + j] * scale;
    bool ok = (mask_mode != 1) || ((j <= row + Wq) && (j >= Wq || bidx > 0));
    sum += ok ? __expf(v - mx) : 0.f;
  }
  red[threadIdx.x] = sum; __syncthreads();
  for (int o = 128; o > 0; o >>= 1) { if (threadIdx.x < o) red[threadIdx.x] += red[threadIdx.x + o]; __syncthreads(); }
  const float inv = 1.f / red[0];
  for (int j = threadIdx.x; j < cols; j += 256) {
    float v = S[base + j] * scale;
    bool ok = (mask_mode != 1) || ((j <= row + Wq) && (j >= Wq || bidx > 0));
    P[base + j] = f2bf(ok ? __expf(v - mx) * inv : 0.f);
  }
}

// ---------------------------------------------------------------------------
// Elementwise / data-movement kernels (grid-stride).
// ---------------------------------------------------------------------------
__global__ void k_f2bf(const float* __restrict__ s, us* __restrict__ d, long long n) {
  GSL(i, n) d[i] = f2bf(s[i]);
}
// transpose-convert: src f32 [K,N] -> dst bf16 [N,K]
__global__ void k_transpose(const float* __restrict__ s, us* __restrict__ d,
                            int K, int N, long long n) {
  GSL(i, n) { long long k = i / N, c = i - k * N; d[c * K + k] = f2bf(s[i]); }
}
__global__ void k_zero_us(us* __restrict__ p, long long n) { GSL(i, n) p[i] = 0; }
__global__ void k_init_state(const float* __restrict__ init, float* __restrict__ st,
                             long long sd, long long n) {
  GSL(i, n) st[i] = init[i % sd];
}
// move chunk between xe[B,L,D] and hbuf[B,W,D]
__global__ void k_block_io(float* __restrict__ xe, float* __restrict__ h,
                           int bidx, int to_h, long long n) {
  const long long WD = (long long)Wc * Dc;
  GSL(i, n) {
    long long b = i / WD, rem = i - b * WD;
    long long src = (b * Lc + (long long)bidx * Wc) * Dc + rem;
    if (to_h) h[i] = xe[src]; else xe[src] = h[i];
  }
}
// kv f32 [B,SEQ,2D] -> k bf16 [B,H,SEQ,DH], vT bf16 [B,H,DH,SEQ]
__global__ void k_split_kv(const float* __restrict__ kv, us* __restrict__ kp,
                           us* __restrict__ vt, int SEQ, long long n) {
  GSL(i, n) {
    long long d = i % DHc, t = i / DHc;
    long long s = t % SEQ; t /= SEQ;
    long long h = t % Hc, b = t / Hc;
    long long kvi = (b * SEQ + s) * (2 * Dc) + h * DHc + d;
    kp[i] = f2bf(kv[kvi]);
    vt[((b * Hc + h) * DHc + d) * SEQ + s] = f2bf(kv[kvi + Dc]);
  }
}
// q f32 [B,SEQ,D] -> bf16 [B,H,SEQ,DH] (no rotary)
__global__ void k_split_plain(const float* __restrict__ q, us* __restrict__ o,
                              int SEQ, long long n) {
  GSL(i, n) {
    long long d = i % DHc, t = i / DHc;
    long long s = t % SEQ; t /= SEQ;
    long long h = t % Hc, b = t / Hc;
    o[i] = f2bf(q[(b * SEQ + s) * Dc + h * DHc + d]);
  }
}
// q f32 [B,W,D] -> rotary(pos=W+p) bf16 [B,H,W,DH] + plain copy
__global__ void k_rot_q(const float* __restrict__ q, us* __restrict__ qr,
                        us* __restrict__ qp, long long n) {
  GSL(i, n) {   // n = B*H*W*(DH/2)
    long long d2 = i % (DHc / 2), t = i / (DHc / 2);
    long long p = t % Wc; t /= Wc;
    long long h = t % Hc, b = t / Hc;
    long long qi = (b * Wc + p) * Dc + h * DHc + 2 * d2;
    float t1 = q[qi], t2 = q[qi + 1];
    float freq = __powf(10000.f, -(2.f * (float)d2) / (float)DHc);
    float sn, cs; __sincosf((float)(Wc + p) * freq, &sn, &cs);
    long long o = ((b * Hc + h) * Wc + p) * DHc + 2 * d2;
    qr[o] = f2bf(t1 * cs - t2 * sn); qr[o + 1] = f2bf(t1 * sn + t2 * cs);
    qp[o] = f2bf(t1);                qp[o + 1] = f2bf(t2);
  }
}
// Kcat[B,H,2W,DH] = rotary(pos=j) of [kmem_prev | kcur]
__global__ void k_kcat(const us* __restrict__ kprev, const us* __restrict__ kcur,
                       us* __restrict__ kc, long long n) {
  GSL(i, n) {   // n = B*H*2W*(DH/2)
    long long d2 = i % (DHc / 2), t = i / (DHc / 2);
    long long j = t % (2 * Wc); t /= (2 * Wc);
    long long h = t % Hc, b = t / Hc;
    const us* src = (j < Wc) ? kprev : kcur;
    long long jj = (j < Wc) ? j : j - Wc;
    long long si = ((b * Hc + h) * Wc + jj) * DHc + 2 * d2;
    float t1 = bf2f(src[si]), t2 = bf2f(src[si + 1]);
    float freq = __powf(10000.f, -(2.f * (float)d2) / (float)DHc);
    float sn, cs; __sincosf((float)j * freq, &sn, &cs);
    long long o = ((b * Hc + h) * (2 * Wc) + j) * DHc + 2 * d2;
    kc[o] = f2bf(t1 * cs - t2 * sn); kc[o + 1] = f2bf(t1 * sn + t2 * cs);
  }
}
// VcatT[B,H,DH,2W] = [vmemT_prev | vT_cur]
__global__ void k_vcat(const us* __restrict__ vprev, const us* __restrict__ vcur,
                       us* __restrict__ vc, long long n) {
  GSL(i, n) {   // n = B*H*DH*2W
    long long j = i % (2 * Wc), t = i / (2 * Wc);
    long long d = t % DHc; t /= DHc;
    long long h = t % Hc, b = t / Hc;
    long long row = (b * Hc + h) * DHc + d;
    vc[i] = (j < Wc) ? vprev[row * Wc + j] : vcur[row * Wc + (j - Wc)];
  }
}

// ---------------------------------------------------------------------------
// Host orchestration
// ---------------------------------------------------------------------------
extern "C" void kernel_launch(void* const* d_in, const int* in_sizes, int n_in,
                              void* d_out, int out_size, void* d_ws, size_t ws_size,
                              hipStream_t stream) {
  (void)in_sizes; (void)n_in; (void)out_size; (void)ws_size;

  // ---- inputs (setup_inputs order, params dict flattened in insertion order)
  const float* X   = (const float*)d_in[0];
  const float* EMB = (const float*)d_in[1];
  const float *ln1w[NLc], *Wqw[NLc], *Wkvw[NLc], *Wow[NLc], *ln2w[NLc], *W1w[NLc], *W2w[NLc];
  for (int l = 0; l < NLc; ++l) {
    int b0 = 2 + l * 7;
    ln1w[l] = (const float*)d_in[b0 + 0];
    Wqw [l] = (const float*)d_in[b0 + 1];
    Wkvw[l] = (const float*)d_in[b0 + 2];
    Wow [l] = (const float*)d_in[b0 + 3];
    ln2w[l] = (const float*)d_in[b0 + 4];
    W1w [l] = (const float*)d_in[b0 + 5];
    W2w [l] = (const float*)d_in[b0 + 6];
  }
  const float* LNS   = (const float*)d_in[30];
  const float* WSKV  = (const float*)d_in[31];
  const float* WSQ   = (const float*)d_in[32];
  const float* WO2   = (const float*)d_in[33];
  const float* WSO   = (const float*)d_in[34];
  const float* WZ    = (const float*)d_in[35];
  const float* BETA  = (const float*)d_in[36];
  const float* INITS = (const float*)d_in[37];
  const float* LNF   = (const float*)d_in[38];
  const float* WLOG  = (const float*)d_in[39];
  float* OUT = (float*)d_out;

  // ---- workspace allocator
  char* wsb = (char*)d_ws; size_t off = 0;
  auto alloc = [&](size_t bytes) -> void* {
    off = (off + 255) & ~(size_t)255; void* p = wsb + off; off += bytes; return p;
  };

  const long long nBL = (long long)Bc * Lc;   // 16384
  const long long nBW = (long long)Bc * Wc;   // 2048
  const long long nBS = (long long)Bc * Sc;   // 2048

  us* xbf  = (us*)alloc(nBL * Ic * 2);
  us* embT = (us*)alloc((size_t)Dc * Ic * 2);
  us *WqT[NLc], *WkvT[NLc], *WoT[NLc], *W1T[NLc], *W2T[NLc];
  for (int l = 0; l < NLc; ++l) {
    WqT [l] = (us*)alloc((size_t)Dc * Dc * 2);
    WkvT[l] = (us*)alloc((size_t)2 * Dc * Dc * 2);
    WoT [l] = (us*)alloc((size_t)Dc * Dc * 2);
    W1T [l] = (us*)alloc((size_t)FFc * Dc * 2);
    W2T [l] = (us*)alloc((size_t)Dc * FFc * 2);
  }
  us* WskvT = (us*)alloc((size_t)2 * Dc * Dc * 2);
  us* WsqT  = (us*)alloc((size_t)Dc * Dc * 2);
  us* Wo2T  = (us*)alloc((size_t)Dc * 2 * Dc * 2);
  us* WsoT  = (us*)alloc((size_t)Dc * 2 * Dc * 2);
  us* WzT   = (us*)alloc((size_t)Dc * Dc * 2);
  us* WlogT = (us*)alloc((size_t)Oc * Dc * 2);

  float* xe   = (float*)alloc(nBL * Dc * 4);     // embeds; reused as per-chunk outputs
  float* hbuf = (float*)alloc(nBW * Dc * 4);
  us*    zb   = (us*)   alloc(nBW * Dc * 2);
  float* qf   = (float*)alloc(nBW * Dc * 4);
  float* kvf  = (float*)alloc(nBW * 2 * Dc * 4);
  us* qrot    = (us*)alloc(nBW * Dc * 2);
  us* qplain  = (us*)alloc(nBW * Dc * 2);
  const long long memN = (long long)NLc * Bc * Hc * Wc * DHc;
  us* kmem[2]  = {(us*)alloc(memN * 2), (us*)alloc(memN * 2)};
  us* vmemT[2] = {(us*)alloc(memN * 2), (us*)alloc(memN * 2)};
  us* Kcat  = (us*)alloc((long long)Bc * Hc * 2 * Wc * DHc * 2);
  us* VcatT = (us*)alloc((long long)Bc * Hc * DHc * 2 * Wc * 2);
  float* Sbuf = (float*)alloc((long long)Bc * Hc * Wc * 2 * Wc * 4);
  us*    Pbuf = (us*)   alloc((long long)Bc * Hc * Wc * 2 * Wc * 2);
  us* attnout = (us*)alloc(nBW * 2 * Dc * 2);
  us* ub      = (us*)alloc(nBW * FFc * 2);
  float* stbuf[2] = {(float*)alloc(nBS * Dc * 4), (float*)alloc(nBS * Dc * 4)};
  us*    szb  = (us*)   alloc(nBS * Dc * 2);
  float* skvf = (float*)alloc(nBS * 2 * Dc * 4);
  us*    skb  = (us*)   alloc(nBS * Dc * 2);
  us*    svT  = (us*)   alloc(nBS * Dc * 2);
  float* sqf  = (float*)alloc(nBS * Dc * 4);
  us*    sqb  = (us*)   alloc(nBS * Dc * 2);
  us* sconcat = (us*)alloc(nBS * 2 * Dc * 2);
  us* soutb   = (us*)alloc(nBS * Dc * 2);

  // ---- launch helpers
  auto ew = [](long long n) { return dim3((unsigned)((n + 255) / 256)); };
  auto tr = [&](const float* s, us* d, int K, int N) {
    long long n = (long long)K * N;
    k_transpose<<<ew(n), 256, 0, stream>>>(s, d, K, N, n);
  };
  auto ln = [&](const float* x, const float* g, us* o, int rows) {
    k_ln<<<dim3(rows), 256, 0, stream>>>(x, g, o, Dc);
  };
  auto sm = [&](int rows, int cols, int mode, int bidx) {
    k_softmax<<<dim3(rows, Bc * Hc), 256, 0, stream>>>(Sbuf, Pbuf, cols, 0.125f, mode, bidx, Wc);
  };
  auto gemm = [&](int epi, const us* A, const us* Bt, float* C, us* Cb,
                  const float* res, const float* stold, const float* beta,
                  int M, int N, int K, int lda, int ldb, int ldc,
                  int nz, int bdiv,
                  long long soA, long long siA, long long soB, long long siB,
                  long long soC, long long siC) {
    GemmArgs a;
    a.A = A; a.Bt = Bt; a.C = C; a.Cb = Cb; a.res = res; a.stold = stold; a.beta = beta;
    a.M = M; a.N = N; a.K = K; a.lda = lda; a.ldb = ldb; a.ldc = ldc; a.bdiv = bdiv;
    a.soA = soA; a.siA = siA; a.soB = soB; a.siB = siB; a.soC = soC; a.siC = siC;
    dim3 g(N / 64, M / 128, nz), b(256);
    switch (epi) {
      case 0: k_gemm<0><<<g, b, 0, stream>>>(a); break;
      case 1: k_gemm<1><<<g, b, 0, stream>>>(a); break;
      case 2: k_gemm<2><<<g, b, 0, stream>>>(a); break;
      case 3: k_gemm<3><<<g, b, 0, stream>>>(a); break;
      case 4: k_gemm<4><<<g, b, 0, stream>>>(a); break;
    }
  };

  // ---- prep: casts, weight transposes, memory init, embedding
  k_f2bf<<<ew(nBL * Ic), 256, 0, stream>>>(X, xbf, nBL * Ic);
  tr(EMB, embT, Ic, Dc);
  for (int l = 0; l < NLc; ++l) {
    tr(Wqw[l],  WqT[l],  Dc, Dc);
    tr(Wkvw[l], WkvT[l], Dc, 2 * Dc);
    tr(Wow[l],  WoT[l],  Dc, Dc);
    tr(W1w[l],  W1T[l],  Dc, FFc);
    tr(W2w[l],  W2T[l],  FFc, Dc);
  }
  tr(WSKV, WskvT, Dc, 2 * Dc);
  tr(WSQ,  WsqT,  Dc, Dc);
  tr(WO2,  Wo2T,  2 * Dc, Dc);
  tr(WSO,  WsoT,  2 * Dc, Dc);
  tr(WZ,   WzT,   Dc, Dc);
  tr(WLOG, WlogT, Dc, Oc);
  k_zero_us<<<ew(memN), 256, 0, stream>>>(kmem[0], memN);
  k_zero_us<<<ew(memN), 256, 0, stream>>>(vmemT[0], memN);
  k_init_state<<<ew(nBS * Dc), 256, 0, stream>>>(INITS, stbuf[0], (long long)Sc * Dc, nBS * Dc);

  // xe = x @ emb   (M=16384, N=512, K=64)
  gemm(0, xbf, embT, xe, nullptr, nullptr, nullptr, nullptr,
       (int)nBL, Dc, Ic, Ic, Ic, Dc, 1, 1, 0, 0, 0, 0, 0, 0);

  const long long hwd = (long long)Hc * Wc * DHc;   // per-batch q/k stride
  const long long hsd = (long long)Hc * Sc * DHc;

  // ---- chunk loop
  for (int bidx = 0; bidx < NBc; ++bidx) {
    const int pv = bidx & 1, nx = 1 - pv;
    k_block_io<<<ew(nBW * Dc), 256, 0, stream>>>(xe, hbuf, bidx, 1, nBW * Dc);

    for (int l = 0; l < NLc; ++l) {
      ln(hbuf, ln1w[l], zb, (int)nBW);
      gemm(0, zb, WqT[l], qf, nullptr, nullptr, nullptr, nullptr,
           (int)nBW, Dc, Dc, Dc, Dc, Dc, 1, 1, 0, 0, 0, 0, 0, 0);
      gemm(0, zb, WkvT[l], kvf, nullptr, nullptr, nullptr, nullptr,
           (int)nBW, 2 * Dc, Dc, Dc, Dc, 2 * Dc, 1, 1, 0, 0, 0, 0, 0, 0);

      us* kcur  = kmem[nx]  + (long long)l * Bc * Hc * Wc * DHc;
      us* vcur  = vmemT[nx] + (long long)l * Bc * Hc * Wc * DHc;
      us* kprev = kmem[pv]  + (long long)l * Bc * Hc * Wc * DHc;
      us* vprev = vmemT[pv] + (long long)l * Bc * Hc * Wc * DHc;

      long long nkv = (long long)Bc * Hc * Wc * DHc;
      k_split_kv<<<ew(nkv), 256, 0, stream>>>(kvf, kcur, vcur, Wc, nkv);
      long long nkc = (long long)Bc * Hc * 2 * Wc * (DHc / 2);
      k_kcat<<<ew(nkc), 256, 0, stream>>>(kprev, kcur, Kcat, nkc);
      long long nvc = (long long)Bc * Hc * DHc * 2 * Wc;
      k_vcat<<<ew(nvc), 256, 0, stream>>>(vprev, vcur, VcatT, nvc);
      long long nrq = (long long)Bc * Hc * Wc * (DHc / 2);
      k_rot_q<<<ew(nrq), 256, 0, stream>>>(qf, qrot, qplain, nrq);

      // self-attn scores: [W,2W] per (b,h)
      gemm(0, qrot, Kcat, Sbuf, nullptr, nullptr, nullptr, nullptr,
           Wc, 2 * Wc, DHc, DHc, DHc, 2 * Wc, Bc * Hc, Hc,
           hwd, (long long)Wc * DHc,
           (long long)Hc * 2 * Wc * DHc, (long long)2 * Wc * DHc,
           (long long)Hc * Wc * 2 * Wc, (long long)Wc * 2 * Wc);
      sm(Wc, 2 * Wc, 1, bidx);
      // self_out -> attnout[:, :D] (merged-head layout)
      gemm(1, Pbuf, VcatT, nullptr, attnout, nullptr, nullptr, nullptr,
           Wc, DHc, 2 * Wc, 2 * Wc, 2 * Wc, 2 * Dc, Bc * Hc, Hc,
           (long long)Hc * Wc * 2 * Wc, (long long)Wc * 2 * Wc,
           (long long)Hc * DHc * 2 * Wc, (long long)DHc * 2 * Wc,
           (long long)Wc * 2 * Dc, (long long)DHc);

      if (l == RECc) {
        ln(stbuf[pv], LNS, szb, (int)nBS);
        gemm(0, szb, WskvT, skvf, nullptr, nullptr, nullptr, nullptr,
             (int)nBS, 2 * Dc, Dc, Dc, Dc, 2 * Dc, 1, 1, 0, 0, 0, 0, 0, 0);
        long long nskv = (long long)Bc * Hc * Sc * DHc;
        k_split_kv<<<ew(nskv), 256, 0, stream>>>(skvf, skb, svT, Sc, nskv);

        // cross: q_plain vs state K/V -> attnout[:, D:]
        gemm(0, qplain, skb, Sbuf, nullptr, nullptr, nullptr, nullptr,
             Wc, Sc, DHc, DHc, DHc, Sc, Bc * Hc, Hc,
             hwd, (long long)Wc * DHc, hsd, (long long)Sc * DHc,
             (long long)Hc * Wc * Sc, (long long)Wc * Sc);
        sm(Wc, Sc, 0, bidx);
        gemm(1, Pbuf, svT, nullptr, attnout + Dc, nullptr, nullptr, nullptr,
             Wc, DHc, Sc, Sc, Sc, 2 * Dc, Bc * Hc, Hc,
             (long long)Hc * Wc * Sc, (long long)Wc * Sc,
             (long long)Hc * DHc * Sc, (long long)DHc * Sc,
             (long long)Wc * 2 * Dc, (long long)DHc);
        // h += concat(self,cross) @ Wo2
        gemm(2, attnout, Wo2T, hbuf, nullptr, hbuf, nullptr, nullptr,
             (int)nBW, Dc, 2 * Dc, 2 * Dc, 2 * Dc, Dc, 1, 1, 0, 0, 0, 0, 0, 0);

        // state queries
        gemm(0, szb, WsqT, sqf, nullptr, nullptr, nullptr, nullptr,
             (int)nBS, Dc, Dc, Dc, Dc, Dc, 1, 1, 0, 0, 0, 0, 0, 0);
        long long nsq = (long long)Bc * Hc * Sc * DHc;
        k_split_plain<<<ew(nsq), 256, 0, stream>>>(sqf, sqb, Sc, nsq);

        // s_self: sq vs state K/V -> sconcat[:, :D]
        gemm(0, sqb, skb, Sbuf, nullptr, nullptr, nullptr, nullptr,
             Sc, Sc, DHc, DHc, DHc, Sc, Bc * Hc, Hc,
             hsd, (long long)Sc * DHc, hsd, (long long)Sc * DHc,
             (long long)Hc * Sc * Sc, (long long)Sc * Sc);
        sm(Sc, Sc, 0, bidx);
        gemm(1, Pbuf, svT, nullptr, sconcat, nullptr, nullptr, nullptr,
             Sc, DHc, Sc, Sc, Sc, 2 * Dc, Bc * Hc, Hc,
             (long long)Hc * Sc * Sc, (long long)Sc * Sc,
             (long long)Hc * DHc * Sc, (long long)DHc * Sc,
             (long long)Sc * 2 * Dc, (long long)DHc);

        // s_cross: sq vs this chunk's (un-rotated) k/v -> sconcat[:, D:]
        gemm(0, sqb, kcur, Sbuf, nullptr, nullptr, nullptr, nullptr,
             Sc, Wc, DHc, DHc, DHc, Wc, Bc * Hc, Hc,
             hsd, (long long)Sc * DHc, hwd, (long long)Wc * DHc,
             (long long)Hc * Sc * Wc, (long long)Sc * Wc);
        sm(Sc, Wc, 0, bidx);
        gemm(1, Pbuf, vcur, nullptr, sconcat + Dc, nullptr, nullptr, nullptr,
             Sc, DHc, Wc, Wc, Wc, 2 * Dc, Bc * Hc, Hc,
             (long long)Hc * Sc * Wc, (long long)Sc * Wc,
             (long long)Hc * DHc * Wc, (long long)DHc * Wc,
             (long long)Sc * 2 * Dc, (long long)DHc);

        // s_out = sconcat @ Wso (bf16), state = ema(state, s_out @ Wz)
        gemm(1, sconcat, WsoT, nullptr, soutb, nullptr, nullptr, nullptr,
             (int)nBS, Dc, 2 * Dc, 2 * Dc, 2 * Dc, Dc, 1, 1, 0, 0, 0, 0, 0, 0);
        gemm(4, soutb, WzT, stbuf[nx], nullptr, nullptr, stbuf[pv], BETA,
             (int)nBS, Dc, Dc, Dc, Dc, Dc, 1, 1, 0, 0, 0, 0, 0, 0);
      } else {
        // h += merge(self_out) @ Wo   (reads only first D cols of attnout)
        gemm(2, attnout, WoT[l], hbuf, nullptr, hbuf, nullptr, nullptr,
             (int)nBW, Dc, Dc, 2 * Dc, Dc, Dc, 1, 1, 0, 0, 0, 0, 0, 0);
      }

      // FF block
      ln(hbuf, ln2w[l], zb, (int)nBW);
      gemm(3, zb, W1T[l], nullptr, ub, nullptr, nullptr, nullptr,
           (int)nBW, FFc, Dc, Dc, Dc, FFc, 1, 1, 0, 0, 0, 0, 0, 0);
      gemm(2, ub, W2T[l], hbuf, nullptr, hbuf, nullptr, nullptr,
           (int)nBW, Dc, FFc, FFc, FFc, Dc, 1, 1, 0, 0, 0, 0, 0, 0);
    }

    k_block_io<<<ew(nBW * Dc), 256, 0, stream>>>(xe, hbuf, bidx, 0, nBW * Dc);
  }

  // ---- final: LN(out) @ Wlogit -> d_out  (Pbuf reused as bf16 LN output)
  ln(xe, LNF, Pbuf, (int)nBL);
  gemm(0, Pbuf, WlogT, OUT, nullptr, nullptr, nullptr, nullptr,
       (int)nBL, Oc, Dc, Dc, Dc, Oc, 1, 1, 0, 0, 0, 0, 0, 0);
}
